// Encoder_89618787598974
// MI455X (gfx1250) — matrix operand order repaired and checked by reading, not compiled, not run on hardware
//
#include <hip/hip_runtime.h>
#include <hip/hip_bf16.h>
#include <stdint.h>

// ---------------------------------------------------------------------------
// Types for CDNA5 WMMA (wave32): V_WMMA_F32_16X16X32_F16
// ---------------------------------------------------------------------------
typedef __attribute__((ext_vector_type(16))) _Float16 v16h;
typedef __attribute__((ext_vector_type(8)))  _Float16 v8h;
typedef __attribute__((ext_vector_type(8)))  float    v8f;

#define NC   200
#define KM   50          // k_mentions
#define LSEQ 1536
#define DDIM 768
#define HEAD 12
#define NPAIR (KM * KM)          // 2500
#define MPAD  2560               // 160 tiles of 16, divisible by BM=4
#define CCOLS (3 * DDIM)         // 2304
#define NREL  57
#define NNOTA 20
#define WPAD  96                 // 57 rel + 20 nota + 19 pad -> 6 N-tiles
#define OUTC  58

__device__ __forceinline__ v16h combine16(v8h lo, v8h hi) {
  return __builtin_shufflevector(lo, hi, 0,1,2,3,4,5,6,7,8,9,10,11,12,13,14,15);
}

__device__ __forceinline__ float waveSum(float v) {
  #pragma unroll
  for (int off = 16; off > 0; off >>= 1) v += __shfl_xor(v, off, 32);
  return v;
}

// ---------------------------------------------------------------------------
// 1) Span mean pooling of sequence rows: embs[NC, DDIM]
// ---------------------------------------------------------------------------
__global__ void span_embs_kernel(const float* __restrict__ seq,
                                 const int* __restrict__ starts,
                                 const int* __restrict__ lens,
                                 float* __restrict__ embs) {
  int c = blockIdx.x;
  int s = starts[c], ln = lens[c];
  float inv = 1.0f / (float)(ln + 1);
  for (int d = threadIdx.x; d < DDIM; d += blockDim.x) {
    float acc = 0.0f;
    for (int j = 0; j <= ln; ++j) acc += seq[(size_t)(s + j) * DDIM + d];
    embs[(size_t)c * DDIM + d] = acc * inv;
  }
}

// ---------------------------------------------------------------------------
// 2) span_max[c] = max over 18 anchor dot-products (one wave per candidate)
// ---------------------------------------------------------------------------
__global__ void span_max_kernel(const float* __restrict__ embs,
                                const float* __restrict__ anchor,
                                float* __restrict__ span_max) {
  int c = blockIdx.x;
  int lane = threadIdx.x;
  const float* e = embs + (size_t)c * DDIM;
  float best = -__builtin_inff();
  for (int a = 0; a < 18; ++a) {
    const float* w = anchor + (size_t)a * DDIM;
    float s = 0.0f;
    for (int d = lane; d < DDIM; d += 32) s += e[d] * w[d];
    s = waveSum(s);
    best = fmaxf(best, s);
  }
  if (lane == 0) span_max[c] = best;
}

// ---------------------------------------------------------------------------
// 3) Top-K (descending, ties -> lowest index) like lax.top_k
// ---------------------------------------------------------------------------
__global__ void topk_kernel(const float* __restrict__ span_max,
                            int* __restrict__ topidx) {
  __shared__ float v[256];
  __shared__ float bv[256];
  __shared__ int   bi[256];
  int t = threadIdx.x;
  v[t] = (t < NC) ? span_max[t] : -__builtin_inff();
  __syncthreads();
  for (int k = 0; k < KM; ++k) {
    bv[t] = v[t]; bi[t] = t;
    __syncthreads();
    for (int off = 128; off > 0; off >>= 1) {
      if (t < off) {
        float ov = bv[t + off]; int oi = bi[t + off];
        if (ov > bv[t] || (ov == bv[t] && oi < bi[t])) { bv[t] = ov; bi[t] = oi; }
      }
      __syncthreads();
    }
    if (t == 0) { topidx[k] = bi[0]; v[bi[0]] = -__builtin_inff(); }
    __syncthreads();
  }
}

// ---------------------------------------------------------------------------
// 4) Gather span-pooled attention rows for the top-K: ea[KM, HEAD, LSEQ]
// ---------------------------------------------------------------------------
__global__ void gather_ea_kernel(const float* __restrict__ attn,
                                 const int* __restrict__ starts,
                                 const int* __restrict__ lens,
                                 const int* __restrict__ topidx,
                                 float* __restrict__ ea) {
  int k = blockIdx.x;
  int c = topidx[k];
  int s = starts[c], ln = lens[c];
  float inv = 1.0f / (float)(ln + 1);
  const int HL = HEAD * LSEQ;
  for (int t = threadIdx.x; t < HL; t += blockDim.x) {
    float acc = 0.0f;
    for (int j = 0; j <= ln; ++j) acc += attn[(size_t)(s + j) * HL + t];
    ea[(size_t)k * HL + t] = acc * inv;
  }
}

// ---------------------------------------------------------------------------
// 5) Per-pair localized attention: a16[pq, l] = normalize_l( sum_h ea_p*ea_q )
//    Pad rows (pq >= NPAIR) are zero-filled for the WMMA GEMM.
// ---------------------------------------------------------------------------
__global__ void pair_attn_kernel(const float* __restrict__ ea,
                                 _Float16* __restrict__ a16) {
  __shared__ float qv[LSEQ];
  __shared__ float red[256];
  int pq = blockIdx.x;
  _Float16* row = a16 + (size_t)pq * LSEQ;
  if (pq >= NPAIR) {
    for (int l = threadIdx.x; l < LSEQ; l += blockDim.x) row[l] = (_Float16)0.0f;
    return;
  }
  int p = pq / KM, q = pq % KM;
  const float* eap = ea + (size_t)p * HEAD * LSEQ;
  const float* eaq = ea + (size_t)q * HEAD * LSEQ;
  float part = 0.0f;
  for (int l = threadIdx.x; l < LSEQ; l += blockDim.x) {
    float s = 0.0f;
    #pragma unroll
    for (int h = 0; h < HEAD; ++h) s += eap[h * LSEQ + l] * eaq[h * LSEQ + l];
    qv[l] = s;
    part += s;
  }
  red[threadIdx.x] = part;
  __syncthreads();
  for (int off = 128; off > 0; off >>= 1) {
    if (threadIdx.x < off) red[threadIdx.x] += red[threadIdx.x + off];
    __syncthreads();
  }
  float inv = 1.0f / red[0];
  for (int l = threadIdx.x; l < LSEQ; l += blockDim.x)
    row[l] = (_Float16)(qv[l] * inv);
}

// ---------------------------------------------------------------------------
// 6) seqT[n, l] = (f16) seq[l, n]   (B operand stored column-major: [DDIM, LSEQ])
// ---------------------------------------------------------------------------
__global__ void seqT_kernel(const float* __restrict__ seq,
                            _Float16* __restrict__ seqT) {
  int i = blockIdx.x * blockDim.x + threadIdx.x;
  if (i >= DDIM * LSEQ) return;
  int n = i / LSEQ, l = i % LSEQ;
  seqT[i] = (_Float16)seq[(size_t)l * DDIM + n];
}

// ---------------------------------------------------------------------------
// 7) Pack [rel(57); nota(20); pad(19)] rows, K=2304, into f16 (col-major B)
// ---------------------------------------------------------------------------
__global__ void weights_kernel(const float* __restrict__ rel,
                               const float* __restrict__ nota,
                               _Float16* __restrict__ w16) {
  int i = blockIdx.x * blockDim.x + threadIdx.x;
  if (i >= WPAD * CCOLS) return;
  int r = i / CCOLS, c = i % CCOLS;
  float v = 0.0f;
  if (r < NREL) v = rel[(size_t)r * CCOLS + c];
  else if (r < NREL + NNOTA) v = nota[(size_t)(r - NREL) * CCOLS + c];
  w16[i] = (_Float16)v;
}

// ---------------------------------------------------------------------------
// 8) concat cols [0,1536): [ee_p ; ee_q], f16. Pad rows fully zeroed.
//    Cols [1536,2304) (= c) are written by the first WMMA GEMM.
// ---------------------------------------------------------------------------
__global__ void concat_ee_kernel(const float* __restrict__ embs,
                                 const int* __restrict__ topidx,
                                 _Float16* __restrict__ concat) {
  int pq = blockIdx.x;
  _Float16* row = concat + (size_t)pq * CCOLS;
  if (pq >= NPAIR) {
    for (int c = threadIdx.x; c < CCOLS; c += blockDim.x) row[c] = (_Float16)0.0f;
    return;
  }
  int p = pq / KM, q = pq % KM;
  const float* ep = embs + (size_t)topidx[p] * DDIM;
  const float* eq = embs + (size_t)topidx[q] * DDIM;
  for (int c = threadIdx.x; c < 2 * DDIM; c += blockDim.x)
    row[c] = (_Float16)((c < DDIM) ? ep[c] : eq[c - DDIM]);
}

// ---------------------------------------------------------------------------
// 9) Register-blocked WMMA GEMM: C[M,N] = A[Mpad,K] (row-major f16) x
//    Bt[Npad,K] (B column-major f16). Each wave computes a BM x BN block of
//    16x16 tiles (BM*BN f32 accumulators), K-step 32. A fragments are reused
//    BN times and B fragments BM times -> ~2.7x less L2 fragment traffic than
//    1x1 tiling. All fragment loads are contiguous 16B. Output either f32
//    (Cf) or f16 (Ch at column offset col0).
// ---------------------------------------------------------------------------
template <int BM, int BN>
__global__ void wmma_gemm_kernel(const _Float16* __restrict__ A,
                                 const _Float16* __restrict__ Bt,
                                 int M, int Mpad, int Npad, int K,
                                 float* __restrict__ Cf,
                                 _Float16* __restrict__ Ch,
                                 int ldc, int col0) {
  const int wid    = (blockIdx.x * blockDim.x + threadIdx.x) >> 5;
  const int tilesN = Npad / (16 * BN);
  const int tilesM = Mpad / (16 * BM);
  const int tm = wid / tilesN;
  const int tn = wid % tilesN;
  if (tm >= tilesM) return;               // wave-uniform exit; EXEC stays full

  const int lane = threadIdx.x & 31;
  const int lm = lane & 15;
  const int hi = lane >> 4;               // 0: lanes 0-15, 1: lanes 16-31

  const _Float16* Arow[BM];
  const _Float16* Brow[BN];
  #pragma unroll
  for (int i = 0; i < BM; ++i)
    Arow[i] = A + (size_t)((tm * BM + i) * 16 + lm) * K;
  #pragma unroll
  for (int j = 0; j < BN; ++j)
    Brow[j] = Bt + (size_t)((tn * BN + j) * 16 + lm) * K;

  v8f acc[BM][BN];
  #pragma unroll
  for (int i = 0; i < BM; ++i)
    #pragma unroll
    for (int j = 0; j < BN; ++j)
      acc[i][j] = (v8f){};

  for (int k0 = 0; k0 < K; k0 += 32) {
    v16h a[BM], b[BN];
    #pragma unroll
    for (int i = 0; i < BM; ++i) {
      // A 16x32 f16 fragment: halves 0..7 -> K=k0+hi*8..+7 ; 8..15 -> +16
      v8h lo = *(const v8h*)(Arow[i] + k0 + hi * 8);
      v8h hh = *(const v8h*)(Arow[i] + k0 + 16 + hi * 8);
      a[i] = combine16(lo, hh);
    }
    #pragma unroll
    for (int j = 0; j < BN; ++j) {
      // B 32x16 f16 fragment: halves 0..7 -> K=k0+hi*8..+7 ; 8..15 -> +16
      v8h lo = *(const v8h*)(Brow[j] + k0 + hi * 8);
      v8h hh = *(const v8h*)(Brow[j] + k0 + 16 + hi * 8);
      b[j] = combine16(lo, hh);
    }
    #pragma unroll
    for (int i = 0; i < BM; ++i)
      #pragma unroll
      for (int j = 0; j < BN; ++j)
        acc[i][j] = __builtin_amdgcn_wmma_f32_16x16x32_f16(
            /*neg_a=*/false, a[i], /*neg_b=*/false, b[j],
            /*c_mod=*/(short)0, acc[i][j],
            /*reuse_a=*/false, /*reuse_b=*/false);
  }

  #pragma unroll
  for (int i = 0; i < BM; ++i) {
    #pragma unroll
    for (int j = 0; j < BN; ++j) {
      const int col = (tn * BN + j) * 16 + lm;
      #pragma unroll
      for (int r = 0; r < 8; ++r) {
        int row = (tm * BM + i) * 16 + hi * 8 + r;  // C/D: VGPR r -> M=r (+8 hi)
        if (row < M) {
          if (Ch) Ch[(size_t)row * ldc + col0 + col] = (_Float16)acc[i][j][r];
          else    Cf[(size_t)row * ldc + col0 + col] = acc[i][j][r];
        }
      }
    }
  }
}

// ---------------------------------------------------------------------------
// 10) Finalize: out[p,q,0] = max(nota logits), out[p,q,1+r] = rel logits,
//     masked by (trigger[p] > 0 && p != q).
// ---------------------------------------------------------------------------
__global__ void finalize_kernel(const float* __restrict__ scores,
                                const int* __restrict__ trig,
                                float* __restrict__ out) {
  int pq = blockIdx.x;
  int p = pq / KM, q = pq % KM;
  bool valid = (trig[p] > 0) && (p != q);
  const float* srow = scores + (size_t)pq * WPAD;
  float* orow = out + (size_t)pq * OUTC;
  int t = threadIdx.x;
  if (t == 0) {
    float m = -__builtin_inff();
    for (int j = NREL; j < NREL + NNOTA; ++j) m = fmaxf(m, srow[j]);
    orow[0] = valid ? m : 0.0f;
  } else if (t <= NREL) {
    orow[t] = valid ? srow[t - 1] : 0.0f;
  }
}

// ---------------------------------------------------------------------------
// Launch
// ---------------------------------------------------------------------------
extern "C" void kernel_launch(void* const* d_in, const int* in_sizes, int n_in,
                              void* d_out, int out_size, void* d_ws, size_t ws_size,
                              hipStream_t stream) {
  const float* seq    = (const float*)d_in[0];   // [1,1536,768]
  const float* attn   = (const float*)d_in[1];   // [1,1536,12,1536]
  const int*   starts = (const int*)d_in[2];     // [1,200]
  const int*   lens   = (const int*)d_in[3];     // [1,200]
  const int*   trig   = (const int*)d_in[4];     // [1,50]
  const float* anchor = (const float*)d_in[5];   // [18,768]
  const float* rel    = (const float*)d_in[6];   // [57,2304]
  const float* nota   = (const float*)d_in[7];   // [20,2304]
  float* out = (float*)d_out;                    // [1,50,50,58]

  uint8_t* ws = (uint8_t*)d_ws;
  size_t off = 0;
  auto carve = [&](size_t bytes) -> uint8_t* {
    uint8_t* p = ws + off;
    off = (off + bytes + 255) & ~(size_t)255;
    return p;
  };
  float*     embs    = (float*)    carve((size_t)NC * DDIM * 4);
  float*     spanmax = (float*)    carve((size_t)NC * 4);
  int*       topidx  = (int*)      carve((size_t)KM * 4);
  float*     ea      = (float*)    carve((size_t)KM * HEAD * LSEQ * 4);
  _Float16*  a16     = (_Float16*) carve((size_t)MPAD * LSEQ * 2);
  _Float16*  seqT    = (_Float16*) carve((size_t)DDIM * LSEQ * 2);
  _Float16*  w16     = (_Float16*) carve((size_t)WPAD * CCOLS * 2);
  _Float16*  concat  = (_Float16*) carve((size_t)MPAD * CCOLS * 2);
  float*     scores  = (float*)    carve((size_t)MPAD * WPAD * 4);
  (void)ws_size; (void)in_sizes; (void)n_in; (void)out_size;

  span_embs_kernel<<<NC, 256, 0, stream>>>(seq, starts, lens, embs);
  span_max_kernel <<<NC, 32, 0, stream>>>(embs, anchor, spanmax);
  topk_kernel     <<<1, 256, 0, stream>>>(spanmax, topidx);
  gather_ea_kernel<<<KM, 256, 0, stream>>>(attn, starts, lens, topidx, ea);
  pair_attn_kernel<<<MPAD, 256, 0, stream>>>(ea, a16);
  seqT_kernel     <<<(DDIM * LSEQ + 255) / 256, 256, 0, stream>>>(seq, seqT);
  weights_kernel  <<<(WPAD * CCOLS + 255) / 256, 256, 0, stream>>>(rel, nota, w16);
  concat_ee_kernel<<<MPAD, 256, 0, stream>>>(embs, topidx, concat);

  // GEMM1: c = a @ seq -> concat[:, 1536:2304] (f16 out)
  // tilesM = 160/4 = 40, tilesN = 48/2 = 24 -> 960 waves
  {
    int waves = (MPAD / (16 * 4)) * (DDIM / (16 * 2));
    int blocks = (waves * 32 + 255) / 256;
    wmma_gemm_kernel<4, 2><<<blocks, 256, 0, stream>>>(a16, seqT,
        NPAIR, MPAD, DDIM, LSEQ, nullptr, concat, CCOLS, 2 * DDIM);
  }
  // GEMM2: scores = concat @ [rel;nota]^T (f32 out)
  // tilesM = 40, tilesN = 96/32 = 3 -> 120 waves
  {
    int waves = (MPAD / (16 * 4)) * (WPAD / (16 * 2));
    int blocks = (waves * 32 + 255) / 256;
    wmma_gemm_kernel<4, 2><<<blocks, 256, 0, stream>>>(concat, w16,
        NPAIR, MPAD, WPAD, CCOLS, scores, nullptr, WPAD, 0);
  }

  finalize_kernel<<<NPAIR, 64, 0, stream>>>(scores, trig, out);
}
